// Attention_20607253086876
// MI455X (gfx1250) — matrix operand order repaired and checked by reading, not compiled, not run on hardware
//
#include <hip/hip_runtime.h>
#include <hip/hip_bf16.h>

#define EMB   1024
#define ADIM  128
#define SEQ   4096
#define BATCH 4

typedef __bf16 bf16_t;
typedef __attribute__((ext_vector_type(16))) __bf16 v16bf;
typedef __attribute__((ext_vector_type(8)))  __bf16 v8bf;
typedef __attribute__((ext_vector_type(8)))  float  v8f;

#define WMMA_BF16(a, b, c) \
  __builtin_amdgcn_wmma_f32_16x16x32_bf16(false, (a), false, (b), (short)0, (c), false, false)

// ---------------------------------------------------------------------------
// Kernel 0: transpose + downconvert weights.  Wt layout: [3][ADIM][EMB] bf16,
// so WMMA B-fragments (K rows contiguous per d-column) are contiguous loads.
// ---------------------------------------------------------------------------
__global__ __launch_bounds__(256) void wprep(const float* __restrict__ Wq,
                                             const float* __restrict__ Wk,
                                             const float* __restrict__ Wv,
                                             bf16_t* __restrict__ Wt) {
  int idx = blockIdx.x * blockDim.x + threadIdx.x;          // 0 .. 3*EMB*ADIM-1
  int w = idx / (EMB * ADIM);
  int rem = idx - w * (EMB * ADIM);
  int e = rem / ADIM;     // row in W  (E dim)
  int d = rem - e * ADIM; // col in W  (D dim)
  const float* W = (w == 0) ? Wq : ((w == 1) ? Wk : Wv);
  Wt[(size_t)w * ADIM * EMB + (size_t)d * EMB + e] = (bf16_t)W[(size_t)e * ADIM + d];
}

// ---------------------------------------------------------------------------
// Helper: build a 16x32 bf16 A-fragment from an f32 row (two 8-elem runs).
// Layout (contiguous): lanes 0-15 K{0..15}, lanes 16-31 K{16..31}.
// ---------------------------------------------------------------------------
__device__ __forceinline__ v16bf a_frag_from_f32(const float* rowk) {
  const float4* p0 = (const float4*)(rowk);        // K base run
  const float4* p1 = (const float4*)(rowk + 8);    // K base+8 run
  float4 f0 = p0[0], f1 = p0[1];
  float4 f2 = p1[0], f3 = p1[1];
  v16bf a;
  a[0]  = (bf16_t)f0.x; a[1]  = (bf16_t)f0.y; a[2]  = (bf16_t)f0.z; a[3]  = (bf16_t)f0.w;
  a[4]  = (bf16_t)f1.x; a[5]  = (bf16_t)f1.y; a[6]  = (bf16_t)f1.z; a[7]  = (bf16_t)f1.w;
  a[8]  = (bf16_t)f2.x; a[9]  = (bf16_t)f2.y; a[10] = (bf16_t)f2.z; a[11] = (bf16_t)f2.w;
  a[12] = (bf16_t)f3.x; a[13] = (bf16_t)f3.y; a[14] = (bf16_t)f3.z; a[15] = (bf16_t)f3.w;
  return a;
}

// ---------------------------------------------------------------------------
// Kernel 1: fused QKV projection.  One wave computes a 32x128 tile (two
// 16-row M-tiles) of one of Q/K/V: each weight B-fragment load feeds TWO
// v_wmma_f32_16x16x32_bf16, halving load/wait density vs one M-tile.
// Q,K stored [B*S, D] bf16; V stored transposed [B][D][S] bf16.
// ---------------------------------------------------------------------------
__global__ __launch_bounds__(128) void qkv_proj(const float* __restrict__ x,
                                                const bf16_t* __restrict__ Wt,
                                                bf16_t* __restrict__ Qb,
                                                bf16_t* __restrict__ Kb,
                                                bf16_t* __restrict__ Vt) {
  const int lane = threadIdx.x & 31;
  const int wid  = threadIdx.x >> 5;
  const int half = lane >> 4;     // 0: lanes 0-15, 1: lanes 16-31
  const int l16  = lane & 15;

  const int row0  = (blockIdx.x * 4 + wid) * 32;  // global row into [B*S]
  const int which = blockIdx.y;                   // 0=Q, 1=K, 2=V

  const bf16_t* W     = Wt + (size_t)which * ADIM * EMB;
  const float*  xrow0 = x + (size_t)(row0 + l16) * EMB;
  const float*  xrow1 = x + (size_t)(row0 + 16 + l16) * EMB;

  v8f acc0[8], acc1[8];
#pragma unroll
  for (int t = 0; t < 8; ++t)
#pragma unroll
    for (int i = 0; i < 8; ++i) { acc0[t][i] = 0.f; acc1[t][i] = 0.f; }

  const int koffA = half * 16;   // A frag run offset per lane half
  const int koffB = half * 16;   // B frag: lanes 0-15 K rows 0..15, 16-31 rows 16..31

  for (int k0 = 0; k0 < EMB; k0 += 32) {
    if (k0 + 32 < EMB) {         // pull next k-step's x lines early
      __builtin_prefetch(xrow0 + k0 + 32 + koffA, 0, 0);
      __builtin_prefetch(xrow1 + k0 + 32 + koffA, 0, 0);
    }
    v16bf a0 = a_frag_from_f32(xrow0 + k0 + koffA);
    v16bf a1 = a_frag_from_f32(xrow1 + k0 + koffA);

#pragma unroll
    for (int t = 0; t < 8; ++t) {
      // B fragment: column n = t*16 + l16 of W, 16 contiguous K values
      const bf16_t* wp = W + (size_t)(t * 16 + l16) * EMB + k0 + koffB;
      v16bf b = *(const v16bf*)wp;
      acc0[t] = WMMA_BF16(a0, b, acc0[t]);   // one weight load, two WMMAs
      acc1[t] = WMMA_BF16(a1, b, acc1[t]);
    }
  }

  // ---- store (C layout: VGPR r -> M = r + half*8, N = l16) ----
  if (which == 2) {
    const int b  = row0 / SEQ;
    const int s0 = row0 - b * SEQ;
    bf16_t* vbase = Vt + (size_t)b * ADIM * SEQ;
#pragma unroll
    for (int t = 0; t < 8; ++t) {
      bf16_t* vp = vbase + (size_t)(t * 16 + l16) * SEQ + s0 + half * 8;
#pragma unroll
      for (int r = 0; r < 8; ++r) { vp[r] = (bf16_t)acc0[t][r]; vp[16 + r] = (bf16_t)acc1[t][r]; }
    }
  } else {
    bf16_t* dst = (which == 0) ? Qb : Kb;
#pragma unroll
    for (int t = 0; t < 8; ++t)
#pragma unroll
      for (int r = 0; r < 8; ++r) {
        dst[(size_t)(row0 + r + half * 8) * ADIM + t * 16 + l16]      = (bf16_t)acc0[t][r];
        dst[(size_t)(row0 + 16 + r + half * 8) * ADIM + t * 16 + l16] = (bf16_t)acc1[t][r];
      }
  }
}

// ---------------------------------------------------------------------------
// Kernel 2: flash attention, one wave per 16-query tile, 32-key tiles.
// ---------------------------------------------------------------------------
__global__ __launch_bounds__(128) void flash_attn(const bf16_t* __restrict__ Qb,
                                                  const bf16_t* __restrict__ Kb,
                                                  const bf16_t* __restrict__ Vt,
                                                  float* __restrict__ out) {
  __shared__ __align__(32) bf16_t Plds[4][16 * 32];

  const int lane = threadIdx.x & 31;
  const int wid  = threadIdx.x >> 5;
  const int half = lane >> 4;
  const int l16  = lane & 15;
  const int koffA = half * 16;
  const int koffB = half * 16;

  const int qt = blockIdx.x * 4 + wid;     // 0 .. B*S/16-1
  const int b  = qt / (SEQ / 16);
  const int q0 = (qt - b * (SEQ / 16)) * 16;

  const bf16_t* Q = Qb + (size_t)b * SEQ * ADIM;
  const bf16_t* K = Kb + (size_t)b * SEQ * ADIM;
  const bf16_t* V = Vt + (size_t)b * ADIM * SEQ;   // [D][S]
  bf16_t* P = Plds[wid];

  // ---- Q A-fragments for the 4 D-chunks, kept in registers ----
  v16bf qa[4];
  const bf16_t* qrow = Q + (size_t)(q0 + l16) * ADIM;
#pragma unroll
  for (int c = 0; c < 4; ++c) {
    v8bf lo = *(const v8bf*)(qrow + c * 32 + koffA);
    v8bf hi = *(const v8bf*)(qrow + c * 32 + 8 + koffA);
    qa[c] = __builtin_shufflevector(lo, hi, 0, 1, 2, 3, 4, 5, 6, 7,
                                            8, 9, 10, 11, 12, 13, 14, 15);
  }

  v8f o[8];
#pragma unroll
  for (int t = 0; t < 8; ++t)
#pragma unroll
    for (int i = 0; i < 8; ++i) o[t][i] = 0.f;
  float m[8], l[8];
#pragma unroll
  for (int r = 0; r < 8; ++r) { m[r] = -3.0e38f; l[r] = 0.f; }

  const float scale = 0.08838834764831845f;   // 128^-0.5

  for (int j0 = 0; j0 <= q0; j0 += 32) {
    const bool t1 = (j0 + 16) <= q0;   // second 16-key sub-tile has valid keys

    if (j0 + 32 <= q0) {               // guarded (in-bounds) K prefetch for next tile
      __builtin_prefetch(K + (size_t)(j0 + 32 + l16) * ADIM + koffB, 0, 0);
      __builtin_prefetch(K + (size_t)(j0 + 48 + l16) * ADIM + koffB, 0, 0);
    }

    // ---- scores: S = Q K^T, two 16x16 N-tiles ----
    v8f s0, s1;
#pragma unroll
    for (int i = 0; i < 8; ++i) { s0[i] = 0.f; s1[i] = 0.f; }
#pragma unroll
    for (int c = 0; c < 4; ++c) {
      v16bf kb0 = *(const v16bf*)(K + (size_t)(j0 + l16) * ADIM + c * 32 + koffB);
      s0 = WMMA_BF16(qa[c], kb0, s0);
    }
    if (t1) {
#pragma unroll
      for (int c = 0; c < 4; ++c) {
        v16bf kb1 = *(const v16bf*)(K + (size_t)(j0 + 16 + l16) * ADIM + c * 32 + koffB);
        s1 = WMMA_BF16(qa[c], kb1, s1);
      }
    }

    // ---- scale + causal mask (C layout: row M = r + half*8, col N = l16) ----
#pragma unroll
    for (int r = 0; r < 8; ++r) {
      const int query = q0 + r + half * 8;
      s0[r] = ((j0 + l16) <= query) ? s0[r] * scale : -3.0e38f;
      s1[r] = (t1 && (j0 + 16 + l16) <= query) ? s1[r] * scale : -3.0e38f;
    }

    // ---- online softmax (row = one VGPR across a 16-lane half) ----
    float alpha[8];
#pragma unroll
    for (int r = 0; r < 8; ++r) {
      float mx = fmaxf(s0[r], s1[r]);
#pragma unroll
      for (int off = 1; off < 16; off <<= 1) mx = fmaxf(mx, __shfl_xor(mx, off, 16));
      const float mnew = fmaxf(m[r], mx);
      alpha[r] = __expf(m[r] - mnew);
      const float p0 = __expf(s0[r] - mnew);
      const float p1 = __expf(s1[r] - mnew);
      float rs = p0 + p1;
#pragma unroll
      for (int off = 1; off < 16; off <<= 1) rs += __shfl_xor(rs, off, 16);
      l[r] = l[r] * alpha[r] + rs;
      m[r] = mnew;
      const int M = r + half * 8;
      P[M * 32 + l16]      = (bf16_t)p0;   // C layout -> LDS row-major 16x32
      P[M * 32 + 16 + l16] = (bf16_t)p1;
    }
#pragma unroll
    for (int t = 0; t < 8; ++t)
#pragma unroll
      for (int r = 0; r < 8; ++r) o[t][r] *= alpha[r];

    // intra-wave cross-lane LDS dependency: drain DS pipe before re-reading
    asm volatile("s_wait_dscnt 0" ::: "memory");

    // ---- P as 16x32 A-fragment ----
    v8bf plo = *(const v8bf*)&P[l16 * 32 + koffA];
    v8bf phi = *(const v8bf*)&P[l16 * 32 + 8 + koffA];
    v16bf pa = __builtin_shufflevector(plo, phi, 0, 1, 2, 3, 4, 5, 6, 7,
                                                 8, 9, 10, 11, 12, 13, 14, 15);

    // ---- O += P V : V^T stored [D][S] so B-fragment is contiguous ----
#pragma unroll
    for (int t = 0; t < 8; ++t) {
      v16bf vb = *(const v16bf*)(V + (size_t)(t * 16 + l16) * SEQ + j0 + koffB);
      o[t] = WMMA_BF16(pa, vb, o[t]);
    }
  }

  // ---- normalize and write out [B,S,D] f32 ----
  float inv[8];
#pragma unroll
  for (int r = 0; r < 8; ++r) inv[r] = 1.f / l[r];
  float* op = out + ((size_t)b * SEQ + q0) * ADIM;
#pragma unroll
  for (int t = 0; t < 8; ++t)
#pragma unroll
    for (int r = 0; r < 8; ++r)
      op[(size_t)(r + half * 8) * ADIM + t * 16 + l16] = o[t][r] * inv[r];
}

// ---------------------------------------------------------------------------
extern "C" void kernel_launch(void* const* d_in, const int* in_sizes, int n_in,
                              void* d_out, int out_size, void* d_ws, size_t ws_size,
                              hipStream_t stream) {
  const float* x  = (const float*)d_in[0];
  const float* Wq = (const float*)d_in[1];
  const float* Wk = (const float*)d_in[2];
  const float* Wv = (const float*)d_in[3];
  float* out = (float*)d_out;

  const size_t QKV_BYTES = (size_t)BATCH * SEQ * ADIM * sizeof(bf16_t); // 4 MiB
  char* ws = (char*)d_ws;
  bf16_t* Qb = (bf16_t*)(ws);
  bf16_t* Kb = (bf16_t*)(ws + QKV_BYTES);
  bf16_t* Vt = (bf16_t*)(ws + 2 * QKV_BYTES);
  bf16_t* Wt = (bf16_t*)(ws + 3 * QKV_BYTES + 8192);  // 8 KiB slack past Vt

  // 0) transpose/convert weights: 3*EMB*ADIM elements
  wprep<<<(3 * EMB * ADIM) / 256, 256, 0, stream>>>(Wq, Wk, Wv, Wt);
  // 1) QKV projection: (B*S/32)/4 blocks of 4 waves, y = which matrix
  qkv_proj<<<dim3((BATCH * SEQ / 32) / 4, 3), 128, 0, stream>>>(x, Wt, Qb, Kb, Vt);
  // 2) flash attention: one wave per 16-query tile
  flash_attn<<<(BATCH * SEQ / 16) / 4, 128, 0, stream>>>(Qb, Kb, Vt, out);
}
